// AF2SmilesTransformer_57784490000566
// MI455X (gfx1250) — matrix operand order repaired and checked by reading, not compiled, not run on hardware
//
#include <hip/hip_runtime.h>
#include <hip/hip_bf16.h>

// ---------------------------------------------------------------------------
// Model constants (from the reference)
// ---------------------------------------------------------------------------
#define B_  32
#define S_  200
#define M_  256
#define D_  1024
#define H_  16
#define F_  2048
#define L_  16
#define V_  128
#define DEMB_ 384
#define DH_ 64

#define NTOK  (B_ * S_)      // 6400 decoder rows
#define NMEM  (B_ * M_)      // 8192 memory rows

// ---------------------------------------------------------------------------
// WMMA types (CDNA5 / gfx1250, wave32)
// ---------------------------------------------------------------------------
typedef __attribute__((ext_vector_type(16))) __bf16 v16bf;
typedef __attribute__((ext_vector_type(8)))  float  v8f;

union BF16Frag {
    v16bf v;
    unsigned short u[16];
    uint4 q[2];
};

// packed f32x2 -> bf16x2 in one VALU op (CDNA5 v_cvt_pk_bf16_f32)
__device__ __forceinline__ unsigned int pk2bf(float a, float b) {
    unsigned int r;
    asm("v_cvt_pk_bf16_f32 %0, %1, %2" : "=v"(r) : "v"(a), "v"(b));
    return r;
}

// generic pointer -> 32-bit LDS byte offset (addrspacecast to AS3)
__device__ __forceinline__ unsigned int lds_off(const void* p) {
    return (unsigned int)(unsigned long long)
        (__attribute__((address_space(3))) const void*)p;
}

// ---------------------------------------------------------------------------
// Generic GEMM: C[Mrows,N] = fmax(A[Mrows,K] @ W[K,N] + bias, lo)
// bias must be a valid pointer (zero vector for "no bias");
// lo = 0.f for ReLU, -3e38f for identity. Branch-free epilogue.
// Requires Mrows%128==0, N%64==0, K%64==0.
// Block = 256 threads = 8 waves. Block tile 128x64, wave tile 32x32.
// BK=64: 8 WMMA + 16 ds_load_b128 per barrier pair.
// ---------------------------------------------------------------------------
#define BM 128
#define BN 64
#define BK 64
#define LDA (BK + 8)   // 72 halves = 144 B row stride (16B-aligned offsets)
#define LDB (BK + 8)

__global__ __launch_bounds__(256)
void gemm_bf16_wmma(const float* __restrict__ A, const float* __restrict__ W,
                    const float* __restrict__ bias, float* __restrict__ C,
                    int Mrows, int N, int K, float lo)
{
    __shared__ __align__(16) unsigned short As[BM][LDA];  // [row][k]   18.0 KB
    __shared__ __align__(16) unsigned short Bs[BN][LDB];  // [n][k]      9.0 KB

    const int tid  = threadIdx.x;
    const int wid  = tid >> 5;        // wave 0..7
    const int lane = tid & 31;
    const int wm   = wid & 3;         // 4 row-waves  * 32 rows = 128
    const int wn   = wid >> 2;        // 2 col-waves  * 32 cols = 64
    const int rowBase = blockIdx.y * BM;
    const int colBase = blockIdx.x * BN;
    const int m  = lane & 15;
    const int hi = lane >> 4;

    v8f acc[2][2] = {};

    // staging assignments
    const int ar  = tid >> 1;          // 0..127  (A row)
    const int ac0 = (tid & 1) * 32;    // 0 / 32  (A col chunk, 32 floats)
    const int bn  = tid & 63;          // 0..63   (B col)
    const int bk0 = (tid >> 6) * 16;   // 0,16,32,48 (B k chunk, 16 floats)

    for (int k0 = 0; k0 < K; k0 += BK) {
        // speculative prefetch of the next k-slab (global_prefetch_b8)
        __builtin_prefetch(A + (size_t)(rowBase + ar) * K + k0 + BK + ac0, 0, 3);
        __builtin_prefetch(W + (size_t)(k0 + BK + bk0) * N + colBase + bn, 0, 3);

        // ---- stage A tile (128x64): 32 consecutive floats per thread
        {
            const float4* src =
                reinterpret_cast<const float4*>(A + (size_t)(rowBase + ar) * K + k0 + ac0);
            alignas(16) unsigned int tmp[16];
            #pragma unroll
            for (int j = 0; j < 8; ++j) {
                const float4 f = src[j];
                tmp[j * 2 + 0] = pk2bf(f.x, f.y);
                tmp[j * 2 + 1] = pk2bf(f.z, f.w);
            }
            uint4* dst = reinterpret_cast<uint4*>(&As[ar][ac0]);
            #pragma unroll
            for (int j = 0; j < 4; ++j)
                dst[j] = *reinterpret_cast<const uint4*>(&tmp[4 * j]);
        }
        // ---- stage B tile (64x64) transposed into Bs[n][k]
        {
            float f[16];
            #pragma unroll
            for (int j = 0; j < 16; ++j)
                f[j] = W[(size_t)(k0 + bk0 + j) * N + colBase + bn];
            alignas(16) unsigned int tmp[8];
            #pragma unroll
            for (int j = 0; j < 8; ++j) tmp[j] = pk2bf(f[2 * j], f[2 * j + 1]);
            uint4* dst = reinterpret_cast<uint4*>(&Bs[bn][bk0]);
            dst[0] = *reinterpret_cast<const uint4*>(&tmp[0]);
            dst[1] = *reinterpret_cast<const uint4*>(&tmp[4]);
        }
        __syncthreads();

        // ---- two 32-deep sub-tiles: 8 WMMAs per barrier pair
        #pragma unroll
        for (int t = 0; t < 2; ++t) {
            BF16Frag a[2], b[2];
            #pragma unroll
            for (int s = 0; s < 2; ++s) {
                const unsigned short* ap = &As[wm * 32 + s * 16 + m][32 * t];
                a[s].q[0] = *reinterpret_cast<const uint4*>(ap + 8 * hi);
                a[s].q[1] = *reinterpret_cast<const uint4*>(ap + 16 + 8 * hi);
                const unsigned short* bp = &Bs[wn * 32 + s * 16 + m][32 * t + 8 * hi];
                b[s].q[0] = *reinterpret_cast<const uint4*>(bp);
                b[s].q[1] = *reinterpret_cast<const uint4*>(bp + 16);
            }
            #pragma unroll
            for (int sm = 0; sm < 2; ++sm)
                #pragma unroll
                for (int sn = 0; sn < 2; ++sn)
                    acc[sm][sn] = __builtin_amdgcn_wmma_f32_16x16x32_bf16(
                        false, a[sm].v, false, b[sn].v, (short)0, acc[sm][sn], false, false);
        }
        __syncthreads();
    }

    // ---- branch-free epilogue: D layout -> row = r + 8*hi, col = lane&15
    const int col0 = colBase + wn * 32 + m;
    const float bv[2] = { bias[col0], bias[col0 + 16] };
    #pragma unroll
    for (int sm = 0; sm < 2; ++sm) {
        #pragma unroll
        for (int sn = 0; sn < 2; ++sn) {
            #pragma unroll
            for (int r = 0; r < 8; ++r) {
                const int row = rowBase + wm * 32 + sm * 16 + r + 8 * hi;
                const float v = fmaxf(acc[sm][sn][r] + bv[sn], lo);
                C[(size_t)row * N + col0 + sn * 16] = v;
            }
        }
    }
}

// ---------------------------------------------------------------------------
// Embedding: h[b,s,:] = tok_emb[x[b,s]] + pos_emb[x[b,s]]   (faithful quirk)
// ---------------------------------------------------------------------------
__global__ __launch_bounds__(256)
void embed_kernel(const int* __restrict__ x, const float* __restrict__ tok,
                  const float* __restrict__ pos, float* __restrict__ h)
{
    const int row = blockIdx.x;             // 0..NTOK-1
    const int t   = x[row];
    const float* te = tok + (size_t)t * D_;
    const float* pe = pos + (size_t)t * D_;
    float* dst = h + (size_t)row * D_;
    for (int d = threadIdx.x; d < D_; d += blockDim.x) dst[d] = te[d] + pe[d];
}

// ---------------------------------------------------------------------------
// Flash-style attention for one (b, head). K/V head-slices are copied to LDS
// with CDNA5 async global->LDS transfers (ASYNCcnt), then per-query online
// softmax. sk is 200 (self) or 256 (cross). causal!=0 => causal+padding mask.
// ---------------------------------------------------------------------------
__global__ __launch_bounds__(256)
void attn_kernel(const float* __restrict__ Q, const float* __restrict__ K,
                 const float* __restrict__ V, const int* __restrict__ x,
                 float* __restrict__ O, int sk, int causal)
{
    extern __shared__ float smem[];
    float* Ks = smem;                 // [sk][64]
    float* Vs = smem + (size_t)sk * DH_;

    const int b    = blockIdx.x >> 4;     // / H_
    const int head = blockIdx.x & 15;
    const int tid  = threadIdx.x;

    // ---- async copy K/V -> LDS, 16 B per transfer per lane (EXEC all-1s here)
    {
        const unsigned ldsK = lds_off(Ks);
        const unsigned ldsV = lds_off(Vs);
        for (int idx = tid * 4; idx < sk * DH_; idx += 256 * 4) {
            const int kr = idx >> 6, d = idx & 63;
            const size_t gofs = ((size_t)(b * sk + kr)) * D_ + head * DH_ + d;
            const unsigned long long gk = (unsigned long long)(K + gofs);
            const unsigned long long gv = (unsigned long long)(V + gofs);
            const unsigned lk = ldsK + (unsigned)idx * 4u;
            const unsigned lv = ldsV + (unsigned)idx * 4u;
            asm volatile("global_load_async_to_lds_b128 %0, %1, off"
                         :: "v"(lk), "v"(gk) : "memory");
            asm volatile("global_load_async_to_lds_b128 %0, %1, off"
                         :: "v"(lv), "v"(gv) : "memory");
        }
        asm volatile("s_wait_asynccnt 0" ::: "memory");
    }
    __syncthreads();

    if (tid >= S_) return;

    float qr[DH_];
    const float* qp = Q + ((size_t)(b * S_ + tid)) * D_ + head * DH_;
    #pragma unroll
    for (int d = 0; d < DH_; ++d) qr[d] = qp[d];

    float acc[DH_];
    #pragma unroll
    for (int d = 0; d < DH_; ++d) acc[d] = 0.f;
    float mrun = -3.0e38f, l = 0.f;
    const float scale = 0.125f;           // 1/sqrt(64)

    for (int k = 0; k < sk; ++k) {
        float s = 0.f;
        const float* kp = Ks + k * DH_;
        #pragma unroll
        for (int d = 0; d < DH_; ++d) s += qr[d] * kp[d];
        s *= scale;
        if (causal) {
            const bool valid = (k <= tid) && (x[b * S_ + k] != 0);
            if (!valid) s = -1e9f;
        }
        const float mn   = fmaxf(mrun, s);
        const float corr = __expf(mrun - mn);
        const float p    = __expf(s - mn);
        l = l * corr + p;
        const float* vp = Vs + k * DH_;
        #pragma unroll
        for (int d = 0; d < DH_; ++d) acc[d] = acc[d] * corr + p * vp[d];
        mrun = mn;
    }
    const float invl = 1.f / l;
    float* op = O + ((size_t)(b * S_ + tid)) * D_ + head * DH_;
    #pragma unroll
    for (int d = 0; d < DH_; ++d) op[d] = acc[d] * invl;
}

// ---------------------------------------------------------------------------
// h = LayerNorm(h + r) * g + b   (in place over h), one block per row (D=1024)
// ---------------------------------------------------------------------------
__global__ __launch_bounds__(256)
void residual_ln_kernel(float* __restrict__ h, const float* __restrict__ r,
                        const float* __restrict__ g, const float* __restrict__ bb)
{
    __shared__ float rs[256], rs2[256];
    const int row = blockIdx.x, t = threadIdx.x;
    float* hp = h + (size_t)row * D_;
    const float* rp = r + (size_t)row * D_;

    float vals[4];
    float s = 0.f, s2 = 0.f;
    #pragma unroll
    for (int j = 0; j < 4; ++j) {
        const float v = hp[t + j * 256] + rp[t + j * 256];
        vals[j] = v; s += v; s2 += v * v;
    }
    rs[t] = s; rs2[t] = s2;
    __syncthreads();
    for (int off = 128; off; off >>= 1) {
        if (t < off) { rs[t] += rs[t + off]; rs2[t] += rs2[t + off]; }
        __syncthreads();
    }
    const float mean = rs[0] * (1.f / D_);
    const float var  = rs2[0] * (1.f / D_) - mean * mean;
    const float inv  = rsqrtf(var + 1e-5f);
    #pragma unroll
    for (int j = 0; j < 4; ++j) {
        const int d = t + j * 256;
        hp[d] = (vals[j] - mean) * inv * g[d] + bb[d];
    }
}

// ---------------------------------------------------------------------------
// log_softmax over V=128 + NLL gather + loss accumulation
// ---------------------------------------------------------------------------
__global__ void zero_vec(float* p, int n) {
    for (int i = threadIdx.x; i < n; i += blockDim.x) p[i] = 0.f;
}

__global__ void zero1(float* p) { *p = 0.f; }

__global__ __launch_bounds__(128)
void logsoftmax_loss_kernel(const float* __restrict__ logits,
                            const int* __restrict__ x,
                            float* __restrict__ logp, float* __restrict__ loss)
{
    __shared__ float red[128];
    const int row = blockIdx.x, t = threadIdx.x;
    const float v = logits[(size_t)row * V_ + t];

    red[t] = v; __syncthreads();
    for (int off = 64; off; off >>= 1) {
        if (t < off) red[t] = fmaxf(red[t], red[t + off]);
        __syncthreads();
    }
    const float mx = red[0];
    __syncthreads();

    const float e = __expf(v - mx);
    red[t] = e; __syncthreads();
    for (int off = 64; off; off >>= 1) {
        if (t < off) red[t] += red[t + off];
        __syncthreads();
    }
    const float lse = __logf(red[0]) + mx;
    const float lp = v - lse;
    logp[(size_t)row * V_ + t] = lp;
    if (t == x[row]) atomicAdd(loss, -lp * (1.0f / S_));
}

// ---------------------------------------------------------------------------
// Host orchestration
// ---------------------------------------------------------------------------
#define NO_RELU (-3.0e38f)

extern "C" void kernel_launch(void* const* d_in, const int* in_sizes, int n_in,
                              void* d_out, int out_size, void* d_ws, size_t ws_size,
                              hipStream_t stream)
{
    (void)in_sizes; (void)n_in; (void)ws_size;

    const int*   x       = (const int*)  d_in[0];
    const float* af_emb  = (const float*)d_in[1];
    const float* tok_emb = (const float*)d_in[2];
    const float* pos_emb = (const float*)d_in[3];
    const float* proj_w  = (const float*)d_in[4];
    const float* proj_b  = (const float*)d_in[5];
    const float* sa_wq   = (const float*)d_in[6];
    const float* sa_wk   = (const float*)d_in[7];
    const float* sa_wv   = (const float*)d_in[8];
    const float* sa_wo   = (const float*)d_in[9];
    const float* ca_wq   = (const float*)d_in[10];
    const float* ca_wk   = (const float*)d_in[11];
    const float* ca_wv   = (const float*)d_in[12];
    const float* ca_wo   = (const float*)d_in[13];
    const float* ffn_w1  = (const float*)d_in[14];
    const float* ffn_b1  = (const float*)d_in[15];
    const float* ffn_w2  = (const float*)d_in[16];
    const float* ffn_b2  = (const float*)d_in[17];
    const float* ln1_g   = (const float*)d_in[18];
    const float* ln1_b   = (const float*)d_in[19];
    const float* ln2_g   = (const float*)d_in[20];
    const float* ln2_b   = (const float*)d_in[21];
    const float* ln3_g   = (const float*)d_in[22];
    const float* ln3_b   = (const float*)d_in[23];
    const float* out_w   = (const float*)d_in[24];
    const float* out_b   = (const float*)d_in[25];

    float* out  = (float*)d_out;
    float* loss = out + (out_size - 1);       // logp is [B,S,V] flat, loss last

    // Workspace layout (floats)
    float* H    = (float*)d_ws;               // [NTOK, D]
    float* MEMB = H    + (size_t)NTOK * D_;   // [NMEM, D]
    float* Qb   = MEMB + (size_t)NMEM * D_;   // [NTOK, D]
    float* Kb   = Qb   + (size_t)NTOK * D_;   // [NMEM, D] (max rows)
    float* Vb   = Kb   + (size_t)NMEM * D_;   // [NMEM, D]
    float* AO   = Vb   + (size_t)NMEM * D_;   // [NTOK, D]
    float* PR   = AO   + (size_t)NTOK * D_;   // [NTOK, D]
    float* F1   = PR   + (size_t)NTOK * D_;   // [NTOK, F]
    float* LG   = F1   + (size_t)NTOK * F_;   // [NTOK, V]
    float* ZB   = LG   + (size_t)NTOK * V_;   // [F] zero bias

    const dim3 blk256(256);
    auto gemm = [&](const float* A, const float* W, const float* bias, float* C,
                    int Mrows, int N, int K, float lo) {
        dim3 grid(N / BN, Mrows / BM);
        gemm_bf16_wmma<<<grid, blk256, 0, stream>>>(A, W, bias, C, Mrows, N, K, lo);
    };

    // zero the shared "no-bias" vector (deterministic each call)
    zero_vec<<<1, blk256, 0, stream>>>(ZB, F_);

    // Embedding + memory projection
    embed_kernel<<<NTOK, blk256, 0, stream>>>(x, tok_emb, pos_emb, H);
    gemm(af_emb, proj_w, proj_b, MEMB, NMEM, D_, DEMB_, NO_RELU);

    const size_t DD = (size_t)D_ * D_;
    const size_t DF = (size_t)D_ * F_;
    const int smem_self  = 2 * S_ * DH_ * (int)sizeof(float);  // 100 KB
    const int smem_cross = 2 * M_ * DH_ * (int)sizeof(float);  // 128 KB

    for (int i = 0; i < L_; ++i) {
        // ---- self-attention
        gemm(H, sa_wq + i * DD, ZB, Qb, NTOK, D_, D_, NO_RELU);
        gemm(H, sa_wk + i * DD, ZB, Kb, NTOK, D_, D_, NO_RELU);
        gemm(H, sa_wv + i * DD, ZB, Vb, NTOK, D_, D_, NO_RELU);
        attn_kernel<<<B_ * H_, blk256, smem_self, stream>>>(Qb, Kb, Vb, x, AO, S_, 1);
        gemm(AO, sa_wo + i * DD, ZB, PR, NTOK, D_, D_, NO_RELU);
        residual_ln_kernel<<<NTOK, blk256, 0, stream>>>(H, PR, ln1_g + i * D_, ln1_b + i * D_);

        // ---- cross-attention
        gemm(H,    ca_wq + i * DD, ZB, Qb, NTOK, D_, D_, NO_RELU);
        gemm(MEMB, ca_wk + i * DD, ZB, Kb, NMEM, D_, D_, NO_RELU);
        gemm(MEMB, ca_wv + i * DD, ZB, Vb, NMEM, D_, D_, NO_RELU);
        attn_kernel<<<B_ * H_, blk256, smem_cross, stream>>>(Qb, Kb, Vb, nullptr, AO, M_, 0);
        gemm(AO, ca_wo + i * DD, ZB, PR, NTOK, D_, D_, NO_RELU);
        residual_ln_kernel<<<NTOK, blk256, 0, stream>>>(H, PR, ln2_g + i * D_, ln2_b + i * D_);

        // ---- FFN
        gemm(H,  ffn_w1 + i * DF, ffn_b1 + i * F_, F1, NTOK, F_, D_, 0.0f);
        gemm(F1, ffn_w2 + i * DF, ffn_b2 + i * D_, PR, NTOK, D_, F_, NO_RELU);
        residual_ln_kernel<<<NTOK, blk256, 0, stream>>>(H, PR, ln3_g + i * D_, ln3_b + i * D_);
    }

    // ---- output head + log-softmax + loss
    gemm(H, out_w, out_b, LG, NTOK, V_, D_, NO_RELU);
    zero1<<<1, 1, 0, stream>>>(loss);
    logsoftmax_loss_kernel<<<NTOK, dim3(128), 0, stream>>>(LG, x, out, loss);
}
